// SinkhornDistance_60292750901876
// MI455X (gfx1250) — matrix operand order repaired and checked by reading, not compiled, not run on hardware
//
#include <hip/hip_runtime.h>
#include <hip/hip_bf16.h>

#define NB   8
#define PP   2048
#define DD   512
#define SK_EPS     0.1f
#define SK_INV_EPS 10.0f
#define SK_ITERS   20

typedef __attribute__((ext_vector_type(16))) __bf16 v16bf;
typedef __attribute__((ext_vector_type(8)))  __bf16 v8bf;
typedef __attribute__((ext_vector_type(8)))  float  v8f;

__device__ __forceinline__ void lse_accum(float& m, float& s, float val) {
  if (val > m) { s = s * __expf(m - val) + 1.0f; m = val; }
  else         { s += __expf(val - m); }
}
__device__ __forceinline__ void lse_merge(float& m, float& s, float m2, float s2) {
  if (m2 > m) { s = s * __expf(m - m2) + s2; m = m2; }
  else        { s += s2 * __expf(m2 - m); }
}

// ---- fill mu/nu outputs, zero u/v scaling vectors ------------------------
__global__ void sk_init(float* mu, float* nu, float* u, float* v) {
  int idx = blockIdx.x * blockDim.x + threadIdx.x;
  if (idx < NB * PP) {
    mu[idx] = 1.0f / PP;
    nu[idx] = 1.0f / PP;
    u[idx]  = 0.0f;
    v[idx]  = 0.0f;
  }
}

// ---- L2-normalize rows of x and y, emit bf16 -----------------------------
__global__ __launch_bounds__(128) void sk_normalize(const float* __restrict__ x,
                                                    const float* __restrict__ y,
                                                    __bf16* __restrict__ xn,
                                                    __bf16* __restrict__ yn) {
  int row = blockIdx.x;
  const float* src;
  __bf16* dst;
  if (row < NB * PP) { src = x + (size_t)row * DD;               dst = xn + (size_t)row * DD; }
  else { int r2 = row - NB * PP; src = y + (size_t)r2 * DD;      dst = yn + (size_t)r2 * DD; }
  float ssq = 0.0f;
  for (int e = threadIdx.x; e < DD; e += 128) { float t = src[e]; ssq += t * t; }
  __shared__ float red[128];
  red[threadIdx.x] = ssq; __syncthreads();
  for (int off = 64; off > 0; off >>= 1) {
    if (threadIdx.x < off) red[threadIdx.x] += red[threadIdx.x + off];
    __syncthreads();
  }
  float inv = rsqrtf(red[0]);
  for (int e = threadIdx.x; e < DD; e += 128) dst[e] = (__bf16)(src[e] * inv);
}

// ---- C = 1 - Xn @ Yn^T via v_wmma_f32_16x16x32_bf16 ----------------------
// One wave computes a 64x64 tile: 4x4 grid of 16x16 WMMA accumulators.
__global__ __launch_bounds__(256) void sk_cost_gemm(const __bf16* __restrict__ Xn,
                                                    const __bf16* __restrict__ Yn,
                                                    float* __restrict__ C) {
  const int wave = threadIdx.x >> 5;
  const int lane = threadIdx.x & 31;
  const int tile = blockIdx.x * 8 + wave;      // 8*(2048/64)^2 = 8192 tiles
  const int b    = tile >> 10;                 // / (32*32)
  const int rem  = tile & 1023;
  const int m0   = (rem >> 5) << 6;
  const int n0   = (rem & 31) << 6;
  const int r    = lane & 15;
  const int hi   = lane >> 4;

  const __bf16* Xb = Xn + (size_t)b * PP * DD;
  const __bf16* Yb = Yn + (size_t)b * PP * DD;

  v8f acc[4][4];
#pragma unroll
  for (int a = 0; a < 4; ++a)
#pragma unroll
    for (int q = 0; q < 4; ++q)
#pragma unroll
      for (int e = 0; e < 8; ++e) acc[a][q][e] = 0.0f;

  for (int k0 = 0; k0 < DD; k0 += 32) {
    v16bf afrag[4], bfrag[4];
    // A (16x32): lane holds row (lane%16); elems 0..7 <- K=k0+16*hi..+7,
    //            elems 8..15 <- K=k0+16*hi+8..+15
#pragma unroll
    for (int a = 0; a < 4; ++a) {
      const __bf16* pa = Xb + (size_t)(m0 + 16 * a + r) * DD + k0 + 16 * hi;
      v8bf lo = *(const v8bf*)pa;
      v8bf hv = *(const v8bf*)(pa + 8);
      v16bf f;
#pragma unroll
      for (int e = 0; e < 8; ++e) { f[e] = lo[e]; f[e + 8] = hv[e]; }
      afrag[a] = f;
    }
    // B (32x16): lane holds col (lane%16); elems 0..15 <- K=k0+16*hi..+15
#pragma unroll
    for (int q = 0; q < 4; ++q) {
      const __bf16* pb = Yb + (size_t)(n0 + 16 * q + r) * DD + k0 + 16 * hi;
      bfrag[q] = *(const v16bf*)pb;
    }
#pragma unroll
    for (int a = 0; a < 4; ++a)
#pragma unroll
      for (int q = 0; q < 4; ++q)
        acc[a][q] = __builtin_amdgcn_wmma_f32_16x16x32_bf16(
            false, afrag[a], false, bfrag[q], (short)0, acc[a][q], false, false);
  }

  float* Cb = C + (size_t)b * PP * PP;
#pragma unroll
  for (int a = 0; a < 4; ++a)
#pragma unroll
    for (int q = 0; q < 4; ++q)
#pragma unroll
      for (int rr = 0; rr < 8; ++rr) {
        int row = m0 + 16 * a + rr + 8 * hi;    // C/D layout: M = r + 8*(lane/16)
        int col = n0 + 16 * q + r;              // N = lane%16
        Cb[(size_t)row * PP + col] = 1.0f - acc[a][q][rr];
      }
}

// ---- u_i = eps*log(mu) - eps*LSE_j((v_j - C_ij)/eps) ---------------------
__global__ __launch_bounds__(256) void sk_row_pass(const float* __restrict__ C,
                                                   const float* __restrict__ v,
                                                   float* __restrict__ u) {
  const int n = blockIdx.x >> 11;
  const int i = blockIdx.x & (PP - 1);
  const float* Crow = C + ((size_t)n * PP + i) * PP;
  const float* vb = v + n * PP;
  float m = -__builtin_inff(), s = 0.0f;
  for (int j = threadIdx.x; j < PP; j += 256)
    lse_accum(m, s, (vb[j] - Crow[j]) * SK_INV_EPS);
  __shared__ float sm[256], ss[256];
  sm[threadIdx.x] = m; ss[threadIdx.x] = s; __syncthreads();
  for (int off = 128; off > 0; off >>= 1) {
    if (threadIdx.x < off) {
      float mm = sm[threadIdx.x], sv = ss[threadIdx.x];
      lse_merge(mm, sv, sm[threadIdx.x + off], ss[threadIdx.x + off]);
      sm[threadIdx.x] = mm; ss[threadIdx.x] = sv;
    }
    __syncthreads();
  }
  if (threadIdx.x == 0) {
    float logmu = __logf(1.0f / PP + 1e-8f);
    u[n * PP + i] = SK_EPS * logmu - SK_EPS * (sm[0] + __logf(ss[0]));
  }
}

// ---- v_j = eps*log(nu) - eps*LSE_i((u_i - C_ij)/eps) ---------------------
// 64 columns per block, 4-way split over rows, coalesced row segments.
__global__ __launch_bounds__(256) void sk_col_pass(const float* __restrict__ C,
                                                   const float* __restrict__ u,
                                                   float* __restrict__ v) {
  const int n = blockIdx.x >> 5;               // / (2048/64)
  const int jBase = (blockIdx.x & 31) << 6;
  const int c  = threadIdx.x & 63;
  const int rg = threadIdx.x >> 6;             // 0..3
  const int j = jBase + c;
  const float* Cb = C + (size_t)n * PP * PP;
  const float* ub = u + n * PP;
  float m = -__builtin_inff(), s = 0.0f;
  for (int i = rg; i < PP; i += 4)
    lse_accum(m, s, (ub[i] - Cb[(size_t)i * PP + j]) * SK_INV_EPS);
  __shared__ float sm[256], ss[256];
  sm[threadIdx.x] = m; ss[threadIdx.x] = s; __syncthreads();
  if (rg == 0) {
    float mm = sm[c], sv = ss[c];
    lse_merge(mm, sv, sm[c + 64],  ss[c + 64]);
    lse_merge(mm, sv, sm[c + 128], ss[c + 128]);
    lse_merge(mm, sv, sm[c + 192], ss[c + 192]);
    float lognu = __logf(1.0f / PP + 1e-8f);
    v[n * PP + j] = SK_EPS * lognu - SK_EPS * (mm + __logf(sv));
  }
}

// ---- pi = exp((-C+u+v)/eps); per-row partial of sum(pi*C) ----------------
__global__ __launch_bounds__(256) void sk_pi(const float* __restrict__ C,
                                             const float* __restrict__ u,
                                             const float* __restrict__ v,
                                             float* __restrict__ pi,
                                             float* __restrict__ rowPartial) {
  const int n = blockIdx.x >> 11;
  const int i = blockIdx.x & (PP - 1);
  const size_t base = ((size_t)n * PP + i) * PP;
  const float ui = u[n * PP + i];
  const float* vb = v + n * PP;
  float local = 0.0f;
  for (int j = threadIdx.x; j < PP; j += 256) {
    float c = C[base + j];
    float p = __expf((ui + vb[j] - c) * SK_INV_EPS);
    pi[base + j] = p;
    local += p * c;
  }
  __shared__ float red[256];
  red[threadIdx.x] = local; __syncthreads();
  for (int off = 128; off > 0; off >>= 1) {
    if (threadIdx.x < off) red[threadIdx.x] += red[threadIdx.x + off];
    __syncthreads();
  }
  if (threadIdx.x == 0) rowPartial[blockIdx.x] = red[0];
}

__global__ __launch_bounds__(256) void sk_cost_reduce(const float* __restrict__ rowPartial,
                                                      float* __restrict__ cost) {
  const int n = blockIdx.x;
  float local = 0.0f;
  for (int i = threadIdx.x; i < PP; i += 256) local += rowPartial[(size_t)n * PP + i];
  __shared__ float red[256];
  red[threadIdx.x] = local; __syncthreads();
  for (int off = 128; off > 0; off >>= 1) {
    if (threadIdx.x < off) red[threadIdx.x] += red[threadIdx.x + off];
    __syncthreads();
  }
  if (threadIdx.x == 0) cost[n] = red[0];
}

extern "C" void kernel_launch(void* const* d_in, const int* in_sizes, int n_in,
                              void* d_out, int out_size, void* d_ws, size_t ws_size,
                              hipStream_t stream) {
  (void)in_sizes; (void)n_in; (void)out_size; (void)ws_size;
  const float* x = (const float*)d_in[0];
  const float* y = (const float*)d_in[1];

  // d_out: [cost(8) | mu(8*2048) | nu(8*2048) | pi(8*2048*2048) | C(8*2048*2048)]
  float* out  = (float*)d_out;
  float* cost = out;
  float* mu   = out + NB;
  float* nu   = mu + (size_t)NB * PP;
  float* pi   = nu + (size_t)NB * PP;
  float* C    = pi + (size_t)NB * PP * PP;

  char* ws = (char*)d_ws;
  __bf16* Xn = (__bf16*)ws;                                      // 16 MB
  __bf16* Yn = (__bf16*)(ws + (size_t)NB * PP * DD * 2);         // 16 MB
  float* u          = (float*)(ws + (size_t)2 * NB * PP * DD * 2);
  float* v          = u + (size_t)NB * PP;
  float* rowPartial = v + (size_t)NB * PP;                       // NB*PP floats

  sk_init<<<(NB * PP + 255) / 256, 256, 0, stream>>>(mu, nu, u, v);
  sk_normalize<<<2 * NB * PP, 128, 0, stream>>>(x, y, Xn, Yn);
  sk_cost_gemm<<<(NB * (PP / 64) * (PP / 64)) / 8, 256, 0, stream>>>(Xn, Yn, C);
  for (int it = 0; it < SK_ITERS; ++it) {
    sk_row_pass<<<NB * PP, 256, 0, stream>>>(C, v, u);
    sk_col_pass<<<NB * (PP / 64), 256, 0, stream>>>(C, u, v);
  }
  sk_pi<<<NB * PP, 256, 0, stream>>>(C, u, v, pi, rowPartial);
  sk_cost_reduce<<<NB, 256, 0, stream>>>(rowPartial, cost);
}
